// MultiHeadAttentionWithRoPE_34746285425442
// MI455X (gfx1250) — matrix operand order repaired and checked by reading, not compiled, not run on hardware
//
#include <hip/hip_runtime.h>

#define B_   4
#define S_   2048
#define DM_  1024
#define NH_  16
#define DK_  64
#define MTOT (B_ * S_)          // 8192 rows
#define SCALE_ 0.125f           // 1/sqrt(64)

typedef __attribute__((ext_vector_type(16))) __bf16 v16bf;
typedef __attribute__((ext_vector_type(8)))  float  v8f;

union Frag16 {
  v16bf v;
  uint4 u[2];
};

__device__ __forceinline__ unsigned short f2bf_raw(float f) {
  unsigned u = __float_as_uint(f);
  unsigned r = u + 0x7FFFu + ((u >> 16) & 1u);   // round-to-nearest-even
  return (unsigned short)(r >> 16);
}

__device__ __forceinline__ __bf16 f2bf(float f) {
  unsigned short h = f2bf_raw(f);
  __bf16 b;
  __builtin_memcpy(&b, &h, 2);
  return b;
}

__device__ __forceinline__ v8f wmma_bf16(v16bf a, v16bf b, v8f c) {
  return __builtin_amdgcn_wmma_f32_16x16x32_bf16(false, a, false, b, (short)0, c,
                                                 false, false);
}

// ---------------------------------------------------------------- convert ---
__global__ void cvt_f32_bf16(const float* __restrict__ x, __bf16* __restrict__ y,
                             int n) {
  int i = blockIdx.x * blockDim.x + threadIdx.x;
  if (i < n) y[i] = f2bf(x[i]);
}

// ------------------------------------------------------------------- GEMM ---
// Y[m,n] = sum_k X[m,k] * W[n,k]   (X: MxK bf16 row-major, W: NxK bf16 row-major)
// One wave computes a 32(M) x 64(N) tile: 2 A-frags x 4 B-frags, 8 WMMA / k-step.
__global__ void __launch_bounds__(256)
gemm_bf16_nt(const __bf16* __restrict__ X, const __bf16* __restrict__ W,
             float* __restrict__ Y, int M, int N, int K) {
  const int lane = threadIdx.x & 31;
  const int wv   = threadIdx.x >> 5;
  const int gw   = blockIdx.x * (blockDim.x >> 5) + wv;
  const int mtiles = M >> 5;
  const int ntiles = N >> 6;
  if (gw >= mtiles * ntiles) return;
  const int m0 = (gw % mtiles) * 32;
  const int n0 = (gw / mtiles) * 64;
  const int mr = lane & 15;
  const int hl = lane >> 4;

  v8f acc[2][4];
#pragma unroll
  for (int i = 0; i < 2; ++i)
#pragma unroll
    for (int j = 0; j < 4; ++j)
#pragma unroll
      for (int r = 0; r < 8; ++r) acc[i][j][r] = 0.f;

  for (int k0 = 0; k0 < K; k0 += 32) {
    Frag16 a[2];
#pragma unroll
    for (int mi = 0; mi < 2; ++mi) {
      size_t base = (size_t)(m0 + mi * 16 + mr) * K + k0;
      a[mi].u[0] = *(const uint4*)(X + base + hl * 16);
      a[mi].u[1] = *(const uint4*)(X + base + hl * 16 + 8);
    }
    Frag16 b[4];
#pragma unroll
    for (int nt = 0; nt < 4; ++nt) {
      size_t base = (size_t)(n0 + nt * 16 + mr) * K + k0 + hl * 16;
      b[nt].u[0] = *(const uint4*)(W + base);
      b[nt].u[1] = *(const uint4*)(W + base + 8);
    }
#pragma unroll
    for (int mi = 0; mi < 2; ++mi)
#pragma unroll
      for (int nt = 0; nt < 4; ++nt)
        acc[mi][nt] = wmma_bf16(a[mi].v, b[nt].v, acc[mi][nt]);
  }

#pragma unroll
  for (int mi = 0; mi < 2; ++mi)
#pragma unroll
    for (int nt = 0; nt < 4; ++nt)
#pragma unroll
      for (int r = 0; r < 8; ++r) {
        int row = m0 + mi * 16 + hl * 8 + r;
        int col = n0 + nt * 16 + mr;
        Y[(size_t)row * N + col] = acc[mi][nt][r];
      }
}

// -------------------------------------------------------- RoPE + repack ----
// Qf/Kf/Vf: f32 [B*S, DM].  Writes:
//   Qh, Kh : bf16 [b][h][s][dk]   (row-major along dk)
//   Vt     : bf16 [b][h][dk][s]   (transposed, row-major along s)
__global__ void rope_pack(const float* __restrict__ Qf, const float* __restrict__ Kf,
                          const float* __restrict__ Vf, __bf16* __restrict__ Qh,
                          __bf16* __restrict__ Kh, __bf16* __restrict__ Vt) {
  unsigned idx = blockIdx.x * blockDim.x + threadIdx.x;  // B*S*NH*32 threads
  if (idx >= (unsigned)(B_ * S_ * NH_ * (DK_ / 2))) return;
  int j = idx & 31;
  int h = (idx >> 5) & 15;
  int s = (idx >> 9) & 2047;
  int b = idx >> 20;

  size_t src = (size_t)(b * S_ + s) * DM_ + h * DK_ + 2 * j;
  float q0 = Qf[src], q1 = Qf[src + 1];
  float k0 = Kf[src], k1 = Kf[src + 1];
  float v0 = Vf[src], v1 = Vf[src + 1];

  float ang = (float)s * __powf(10000.f, -(float)(2 * j) * (1.f / 64.f));
  float sn, cs;
  __sincosf(ang, &sn, &cs);

  unsigned qp = (unsigned)f2bf_raw(q0 * cs - q1 * sn) |
                ((unsigned)f2bf_raw(q0 * sn + q1 * cs) << 16);
  unsigned kp = (unsigned)f2bf_raw(k0 * cs - k1 * sn) |
                ((unsigned)f2bf_raw(k0 * sn + k1 * cs) << 16);

  size_t dst = ((size_t)(b * NH_ + h) * S_ + s) * DK_ + 2 * j;
  *(unsigned*)((char*)Qh + dst * 2) = qp;
  *(unsigned*)((char*)Kh + dst * 2) = kp;

  size_t dv = ((size_t)(b * NH_ + h) * DK_ + 2 * j) * S_ + s;
  Vt[dv] = f2bf(v0);
  Vt[dv + S_] = f2bf(v1);
}

// ------------------------------------------------------- flash attention ---
// One wave per (b, h, 16-row q-tile). 32-key tiles, online softmax,
// O accumulated over 4 d-tiles, stored bf16 into [b*S+q, h*64+d] row-major.
__global__ void __launch_bounds__(128)
flash_attn(const __bf16* __restrict__ Qh, const __bf16* __restrict__ Kh,
           const __bf16* __restrict__ Vt, __bf16* __restrict__ O) {
  __shared__ __align__(16) __bf16 Plds[4][16][32];
  const int lane = threadIdx.x & 31;
  const int wv   = threadIdx.x >> 5;
  const int gw   = blockIdx.x * 4 + wv;
  const int qt = gw & 127;
  const int h  = (gw >> 7) & 15;
  const int b  = gw >> 11;
  const int qs = qt * 16;
  const int n  = lane & 15;
  const int hl = lane >> 4;

  const size_t baseQK = (size_t)(b * NH_ + h) * S_ * DK_;
  const size_t baseV  = (size_t)(b * NH_ + h) * DK_ * S_;

  Frag16 qa[2];
  {
    size_t r = baseQK + (size_t)(qs + n) * DK_;
    qa[0].u[0] = *(const uint4*)(Qh + r + hl * 16);
    qa[0].u[1] = *(const uint4*)(Qh + r + 8 + hl * 16);
    qa[1].u[0] = *(const uint4*)(Qh + r + 32 + hl * 16);
    qa[1].u[1] = *(const uint4*)(Qh + r + 40 + hl * 16);
  }

  v8f o[4];
  float mrow[8], lrow[8];
#pragma unroll
  for (int dt = 0; dt < 4; ++dt)
#pragma unroll
    for (int r = 0; r < 8; ++r) o[dt][r] = 0.f;
#pragma unroll
  for (int r = 0; r < 8; ++r) { mrow[r] = -1e30f; lrow[r] = 0.f; }

  const int nkt = (qt >> 1) + 1;
  for (int kt = 0; kt < nkt; ++kt) {
    const int k0 = kt * 32;
    v8f s0, s1;
    {
      v8f z;
#pragma unroll
      for (int r = 0; r < 8; ++r) z[r] = 0.f;
      Frag16 kb;
      size_t rk = baseQK + (size_t)(k0 + n) * DK_ + hl * 16;
      kb.u[0] = *(const uint4*)(Kh + rk);
      kb.u[1] = *(const uint4*)(Kh + rk + 8);
      s0 = wmma_bf16(qa[0].v, kb.v, z);
      kb.u[0] = *(const uint4*)(Kh + rk + 32);
      kb.u[1] = *(const uint4*)(Kh + rk + 40);
      s0 = wmma_bf16(qa[1].v, kb.v, s0);

      size_t rk2 = baseQK + (size_t)(k0 + 16 + n) * DK_ + hl * 16;
      kb.u[0] = *(const uint4*)(Kh + rk2);
      kb.u[1] = *(const uint4*)(Kh + rk2 + 8);
      s1 = wmma_bf16(qa[0].v, kb.v, z);
      kb.u[0] = *(const uint4*)(Kh + rk2 + 32);
      kb.u[1] = *(const uint4*)(Kh + rk2 + 40);
      s1 = wmma_bf16(qa[1].v, kb.v, s1);
    }

    const bool needmask = (k0 + 31) > qs;
    float p0[8], p1[8], alpha[8];
#pragma unroll
    for (int r = 0; r < 8; ++r) {
      float a_ = s0[r] * SCALE_;
      float b_ = s1[r] * SCALE_;
      if (needmask) {
        int q = qs + hl * 8 + r;
        if (k0 + n > q) a_ = -1e30f;
        if (k0 + 16 + n > q) b_ = -1e30f;
      }
      float rm = fmaxf(a_, b_);
#pragma unroll
      for (int off = 1; off < 16; off <<= 1)
        rm = fmaxf(rm, __shfl_xor(rm, off, 32));
      float nm = fmaxf(mrow[r], rm);
      float al = __expf(mrow[r] - nm);
      float e0 = __expf(a_ - nm);
      float e1 = __expf(b_ - nm);
      float rs = e0 + e1;
#pragma unroll
      for (int off = 1; off < 16; off <<= 1)
        rs += __shfl_xor(rs, off, 32);
      lrow[r] = lrow[r] * al + rs;
      mrow[r] = nm;
      alpha[r] = al;
      p0[r] = e0;
      p1[r] = e1;
    }
#pragma unroll
    for (int dt = 0; dt < 4; ++dt)
#pragma unroll
      for (int r = 0; r < 8; ++r) o[dt][r] *= alpha[r];

    // C-layout -> A-layout relayout of P through LDS (in-order per wave)
#pragma unroll
    for (int r = 0; r < 8; ++r) {
      int row = hl * 8 + r;
      Plds[wv][row][n]      = f2bf(p0[r]);
      Plds[wv][row][16 + n] = f2bf(p1[r]);
    }
    asm volatile("s_wait_dscnt 0" ::: "memory");
    Frag16 pa;
    pa.u[0] = *(const uint4*)&Plds[wv][n][hl * 16];
    pa.u[1] = *(const uint4*)&Plds[wv][n][hl * 16 + 8];

#pragma unroll
    for (int dt = 0; dt < 4; ++dt) {
      Frag16 vb;
      size_t rv = baseV + (size_t)(dt * 16 + n) * S_ + k0 + hl * 16;
      vb.u[0] = *(const uint4*)(Vt + rv);
      vb.u[1] = *(const uint4*)(Vt + rv + 8);
      o[dt] = wmma_bf16(pa.v, vb.v, o[dt]);
    }
  }

  // normalize and store O as bf16 rows of the (B*S, DM) matrix
  size_t ob = (size_t)(b * S_ + qs) * DM_ + h * DK_;
#pragma unroll
  for (int r = 0; r < 8; ++r) {
    float inv = 1.f / lrow[r];
    int q = hl * 8 + r;
#pragma unroll
    for (int dt = 0; dt < 4; ++dt)
      O[ob + (size_t)q * DM_ + dt * 16 + n] = f2bf(o[dt][r] * inv);
  }
}

// ------------------------------------------------------------------ launch ---
extern "C" void kernel_launch(void* const* d_in, const int* in_sizes, int n_in,
                              void* d_out, int out_size, void* d_ws, size_t ws_size,
                              hipStream_t stream) {
  const float* x  = (const float*)d_in[0];
  const float* Wq = (const float*)d_in[1];
  const float* Wk = (const float*)d_in[2];
  const float* Wv = (const float*)d_in[3];
  const float* Wo = (const float*)d_in[4];
  float* out = (float*)d_out;

  const size_t MN = (size_t)MTOT * DM_;   // 8388608 elements
  const size_t WN = (size_t)DM_ * DM_;    // 1048576 elements

  char* ws = (char*)d_ws;
  size_t off = 0;
  auto alloc = [&](size_t bytes) -> void* {
    void* p = ws + off;
    off += (bytes + 255) & ~(size_t)255;
    return p;
  };
  __bf16* xb  = (__bf16*)alloc(MN * 2);
  __bf16* wqb = (__bf16*)alloc(WN * 2);
  __bf16* wkb = (__bf16*)alloc(WN * 2);
  __bf16* wvb = (__bf16*)alloc(WN * 2);
  __bf16* wob = (__bf16*)alloc(WN * 2);
  float*  Qf  = (float*)alloc(MN * 4);
  float*  Kf  = (float*)alloc(MN * 4);
  float*  Vf  = (float*)alloc(MN * 4);
  __bf16* Qh  = (__bf16*)alloc(MN * 2);
  __bf16* Kh  = (__bf16*)alloc(MN * 2);
  __bf16* Vt  = (__bf16*)alloc(MN * 2);
  __bf16* Obf = (__bf16*)Qf;  // Qf dead after rope_pack; reuse for attention out

  // 1) convert inputs to bf16
  cvt_f32_bf16<<<(int)((MN + 255) / 256), 256, 0, stream>>>(x, xb, (int)MN);
  cvt_f32_bf16<<<(int)((WN + 255) / 256), 256, 0, stream>>>(Wq, wqb, (int)WN);
  cvt_f32_bf16<<<(int)((WN + 255) / 256), 256, 0, stream>>>(Wk, wkb, (int)WN);
  cvt_f32_bf16<<<(int)((WN + 255) / 256), 256, 0, stream>>>(Wv, wvb, (int)WN);
  cvt_f32_bf16<<<(int)((WN + 255) / 256), 256, 0, stream>>>(Wo, wob, (int)WN);

  // 2) QKV projections: (8192x1024) @ (1024x1024)^T
  const int gemm_blocks = (MTOT / 32) * (DM_ / 64) / 8;  // 512
  gemm_bf16_nt<<<gemm_blocks, 256, 0, stream>>>(xb, wqb, Qf, MTOT, DM_, DM_);
  gemm_bf16_nt<<<gemm_blocks, 256, 0, stream>>>(xb, wkb, Kf, MTOT, DM_, DM_);
  gemm_bf16_nt<<<gemm_blocks, 256, 0, stream>>>(xb, wvb, Vf, MTOT, DM_, DM_);

  // 3) RoPE on Q,K + pack to bf16 head layouts; V transposed
  const int rope_threads = B_ * S_ * NH_ * (DK_ / 2);  // 4194304
  rope_pack<<<rope_threads / 256, 256, 0, stream>>>(Qf, Kf, Vf, Qh, Kh, Vt);

  // 4) flash attention: one wave per (b,h,q-tile) = 8192 waves, 4 waves/block
  flash_attn<<<(B_ * NH_ * (S_ / 16)) / 4, 128, 0, stream>>>(Qh, Kh, Vt, Obf);

  // 5) output projection into f32 d_out
  gemm_bf16_nt<<<gemm_blocks, 256, 0, stream>>>(Obf, wob, out, MTOT, DM_, DM_);

  (void)in_sizes; (void)n_in; (void)out_size; (void)ws_size;
}
